// PolymorphicAttentionOrchestrator_16673063043645
// MI455X (gfx1250) — matrix operand order repaired and checked by reading, not compiled, not run on hardware
//
#include <hip/hip_runtime.h>
#include <hip/hip_bf16.h>

// ---------------------------------------------------------------------------
// PolymorphicAttentionOrchestrator for MI455X (gfx1250, wave32, WMMA)
// B=2, S=2048, D=1024, H=8, hd=128
// ---------------------------------------------------------------------------

#define S_    2048
#define D_    1024
#define H_    8
#define HD_   128
#define B_    2
#define BH_   (B_*H_)          // 16
#define ROWS_ (B_*S_)          // 4096 token rows
#define NROW_ (BH_*S_)         // 32768 (b,h,q) rows

typedef __attribute__((ext_vector_type(16))) __bf16 v16bf;
typedef __attribute__((ext_vector_type(8)))  __bf16 v8bf;
typedef __attribute__((ext_vector_type(4)))  __bf16 v4bf;
typedef __attribute__((ext_vector_type(8)))  float  v8f;

union AF {            // WMMA 16x16x32 bf16 operand (8 VGPRs / lane)
    v16bf v;
    v8bf  h[2];
    __bf16 e[16];
};

// ---------------------------------------------------------------------------
// f32 -> bf16 convert (vectorized)
// ---------------------------------------------------------------------------
__global__ void k_cvt(const float4* __restrict__ in, __bf16* __restrict__ out, int n4) {
    int i = blockIdx.x * 256 + threadIdx.x;
    if (i < n4) {
        float4 v = in[i];
        v4bf o;
        o[0] = (__bf16)v.x; o[1] = (__bf16)v.y; o[2] = (__bf16)v.z; o[3] = (__bf16)v.w;
        *(v4bf*)&out[i * 4] = o;
    }
}

// ---------------------------------------------------------------------------
// rinv[q,k] = 1 / (dist(q,k)^2 + 1e-6), curvature-warped distance
// ---------------------------------------------------------------------------
__global__ void k_rinv(const float* __restrict__ pos, float* __restrict__ rinv) {
    int idx = blockIdx.x * 256 + threadIdx.x;      // S*S total
    int q = idx >> 11, k = idx & (S_ - 1);
    float s = 0.0f;
#pragma unroll
    for (int j = 0; j < 8; ++j) {
        float d = pos[q * 8 + j] - pos[k * 8 + j];
        s += d * d;
    }
    float dist = sqrtf(s + 1e-12f);
    dist = dist * (1.0f + 0.15f * __sinf(dist));
    dist = fmaxf(dist, 0.0f);
    rinv[idx] = 1.0f / (dist * dist + 1e-6f);
}

// ---------------------------------------------------------------------------
// masses[b,h,s] = sigmoid( x[b,s,h,:] . mass_w[h,:] )   (one wave per row)
// ---------------------------------------------------------------------------
__global__ void k_masses(const float* __restrict__ x, const float* __restrict__ mw,
                         float* __restrict__ masses) {
    int lane = threadIdx.x & 31, wv = threadIdx.x >> 5;
    int r = blockIdx.x * 8 + wv;                   // [0, NROW_)
    int b = r >> 14, rem = r & 16383, h = rem >> 11, s = rem & (S_ - 1);
    const float4 xv = *(const float4*)&x[(size_t)(b * S_ + s) * D_ + h * HD_ + lane * 4];
    const float4 wv4 = *(const float4*)&mw[h * HD_ + lane * 4];
    float t = xv.x * wv4.x + xv.y * wv4.y + xv.z * wv4.z + xv.w * wv4.w;
#pragma unroll
    for (int off = 16; off; off >>= 1) t += __shfl_xor(t, off, 32);
    if (lane == 0) masses[r] = 1.0f / (1.0f + __expf(-t));
}

// ---------------------------------------------------------------------------
// Row stats: rowmax[r] = max_k force, rowisum[r] = 1/sum_k exp(force-max)
// one wave per (b,h,q) row; forces held in registers between the two passes
// ---------------------------------------------------------------------------
__global__ void k_stats(const float* __restrict__ rinv, const float* __restrict__ masses,
                        float* __restrict__ rowmax, float* __restrict__ rowisum) {
    int lane = threadIdx.x & 31, wv = threadIdx.x >> 5;
    int r = blockIdx.x * 8 + wv;                   // [0, NROW_)
    int bh = r >> 11, q = r & (S_ - 1);
    float mq = masses[r];
    const float* rq = rinv + (size_t)q * S_;
    const float* mk = masses + bh * S_;
    float4 fs[16];
    float fm = -1e30f;
#pragma unroll
    for (int j = 0; j < 16; ++j) {
        int k = j * 128 + lane * 4;
        float4 rv = *(const float4*)&rq[k];
        float4 mv = *(const float4*)&mk[k];
        float4 f;
        f.x = fminf(mq * mv.x * rv.x, 50.0f);
        f.y = fminf(mq * mv.y * rv.y, 50.0f);
        f.z = fminf(mq * mv.z * rv.z, 50.0f);
        f.w = fminf(mq * mv.w * rv.w, 50.0f);
        fs[j] = f;
        fm = fmaxf(fm, fmaxf(fmaxf(f.x, f.y), fmaxf(f.z, f.w)));
    }
#pragma unroll
    for (int off = 16; off; off >>= 1) fm = fmaxf(fm, __shfl_xor(fm, off, 32));
    float s = 0.0f;
#pragma unroll
    for (int j = 0; j < 16; ++j)
        s += __expf(fs[j].x - fm) + __expf(fs[j].y - fm) +
             __expf(fs[j].z - fm) + __expf(fs[j].w - fm);
#pragma unroll
    for (int off = 16; off; off >>= 1) s += __shfl_xor(s, off, 32);
    if (lane == 0) { rowmax[r] = fm; rowisum[r] = 1.0f / s; }
}

// ---------------------------------------------------------------------------
// bf16 WMMA GEMM: C(4096x1024) = A(4096x1024) * W(1024x1024)^T
// MODE 0: write bf16 permuted to V layout [b][h][k][d]   (V-projection)
// MODE 1: write f32 row-major to out                     (output projection)
// block = 256 threads (8 waves), tile 128M x 128N, Ktile 128 in LDS
// ---------------------------------------------------------------------------
template <int MODE>
__global__ void k_gemm(const __bf16* __restrict__ A, const __bf16* __restrict__ W,
                       void* __restrict__ Cout) {
    __shared__ __bf16 Wt[128][136];                // [n][k] padded (+8)
    int tid = threadIdx.x, lane = tid & 31, wv = tid >> 5;
    int mb = blockIdx.x >> 3, nb = blockIdx.x & 7;
    int mbase = mb * 128, nbase = nb * 128;
    int row = mbase + wv * 16 + (lane & 15);       // A-fragment row (M = lane&15)
    v8f acc[8];
#pragma unroll
    for (int t = 0; t < 8; ++t) acc[t] = (v8f)(0.0f);

    for (int km = 0; km < D_; km += 128) {
        __syncthreads();
        for (int idx = tid; idx < 128 * 16; idx += 256) {
            int n = idx >> 4, kc = idx & 15;
            *(v8bf*)&Wt[n][kc * 8] =
                *(const v8bf*)&W[(size_t)(nbase + n) * D_ + km + kc * 8];
        }
        __syncthreads();
#pragma unroll
        for (int ks = 0; ks < 128; ks += 32) {
            int klo = km + ks + ((lane >> 4) << 3);       // A: +8 for upper half-wave
            AF a;
            a.h[0] = *(const v8bf*)&A[(size_t)row * D_ + klo];       // K +0..7
            a.h[1] = *(const v8bf*)&A[(size_t)row * D_ + klo + 16];  // K +16..23
            int koff = ks + ((lane >> 4) << 3);           // B: +8 for upper half-wave
#pragma unroll
            for (int t = 0; t < 8; ++t) {
                int col = t * 16 + (lane & 15);
                AF bf;
                bf.h[0] = *(const v8bf*)&Wt[col][koff];
                bf.h[1] = *(const v8bf*)&Wt[col][koff + 16];
                acc[t] = __builtin_amdgcn_wmma_f32_16x16x32_bf16(
                    false, a.v, false, bf.v, (short)0, acc[t], false, false);
            }
        }
    }
    int n = lane & 15, mhi = (lane >> 4) << 3;
#pragma unroll
    for (int t = 0; t < 8; ++t) {
#pragma unroll
        for (int r = 0; r < 8; ++r) {
            int i = mbase + wv * 16 + mhi + r;            // global token row
            int j = nbase + t * 16 + n;                   // global output col
            if (MODE == 0) {
                int b = i >> 11, s = i & (S_ - 1), h = j >> 7, d = j & (HD_ - 1);
                ((__bf16*)Cout)[((size_t)((b << 3) + h) * S_ + s) * HD_ + d] =
                    (__bf16)acc[t][r];
            } else {
                ((float*)Cout)[(size_t)i * D_ + j] = acc[t][r];
            }
        }
    }
}

// ---------------------------------------------------------------------------
// Fused attention: P generated in registers (exp + normalize), P @ V via WMMA
// block = 256 threads (8 waves), handles one (b,h) and 128 q rows
// V tile (128 k x 128 d) staged TRANSPOSED in LDS -> B-frags are 2x b128 reads
// ---------------------------------------------------------------------------
__global__ void k_attn(const float* __restrict__ rinv, const float* __restrict__ masses,
                       const float* __restrict__ rowmax, const float* __restrict__ rowisum,
                       const __bf16* __restrict__ Vbf, __bf16* __restrict__ Hbf) {
    __shared__ __bf16 Vt[128][136];                // [d][k] padded (+8)
    __shared__ float  mks[128];
    int tid = threadIdx.x, lane = tid & 31, wv = tid >> 5;
    int bh = blockIdx.x >> 4, qt = blockIdx.x & 15;
    int b = bh >> 3, h = bh & 7;
    int q = qt * 128 + wv * 16 + (lane & 15);      // A-fragment row
    int rrow = bh * S_ + q;
    float mq = masses[rrow], rmax = rowmax[rrow], risum = rowisum[rrow];
    const float* rq = rinv + (size_t)q * S_;
    v8f acc[8];
#pragma unroll
    for (int t = 0; t < 8; ++t) acc[t] = (v8f)(0.0f);

    for (int km = 0; km < S_; km += 128) {
        __syncthreads();
        // load V tile [k][d] -> transpose into Vt[d][k]
        for (int idx = tid; idx < 128 * 16; idx += 256) {
            int k = idx >> 4, dc = idx & 15;
            v8bf g = *(const v8bf*)&Vbf[((size_t)(bh * S_ + km + k)) * HD_ + dc * 8];
#pragma unroll
            for (int j = 0; j < 8; ++j) Vt[dc * 8 + j][k] = g[j];
        }
        if (tid < 128) mks[tid] = masses[bh * S_ + km + tid];
        __syncthreads();
#pragma unroll
        for (int ks = 0; ks < 128; ks += 32) {
            int kh  = (lane >> 4) << 3;            // 0 or 8
            int klo = km + ks + kh;
            float rv[16];
            *(float4*)&rv[0]  = *(const float4*)&rq[klo];
            *(float4*)&rv[4]  = *(const float4*)&rq[klo + 4];
            *(float4*)&rv[8]  = *(const float4*)&rq[klo + 16];
            *(float4*)&rv[12] = *(const float4*)&rq[klo + 20];
            int kls = ks + kh;
            AF a;                                   // P fragment in bf16
#pragma unroll
            for (int e = 0; e < 16; ++e) {
                int kk = kls + (e < 8 ? e : e + 8); // matches A K-layout
                float f = fminf(mq * mks[kk] * rv[e], 50.0f);
                a.e[e] = (__bf16)(__expf(f - rmax) * risum);
            }
            int koff = ks + ((lane >> 4) << 3);
#pragma unroll
            for (int t = 0; t < 8; ++t) {
                int col = t * 16 + (lane & 15);
                AF bf;
                bf.h[0] = *(const v8bf*)&Vt[col][koff];
                bf.h[1] = *(const v8bf*)&Vt[col][koff + 16];
                acc[t] = __builtin_amdgcn_wmma_f32_16x16x32_bf16(
                    false, a.v, false, bf.v, (short)0, acc[t], false, false);
            }
        }
    }
    // write head_out as bf16 rows [b*S+q][h*128+d] for the final GEMM
    int n = lane & 15, mhi = (lane >> 4) << 3;
#pragma unroll
    for (int t = 0; t < 8; ++t) {
#pragma unroll
        for (int r = 0; r < 8; ++r) {
            int qq = qt * 128 + wv * 16 + mhi + r;
            Hbf[((size_t)(b * S_ + qq)) * D_ + h * HD_ + t * 16 + n] = (__bf16)acc[t][r];
        }
    }
}

// ---------------------------------------------------------------------------
// Launcher
// ---------------------------------------------------------------------------
extern "C" void kernel_launch(void* const* d_in, const int* in_sizes, int n_in,
                              void* d_out, int out_size, void* d_ws, size_t ws_size,
                              hipStream_t stream) {
    const float* x      = (const float*)d_in[0];   // (B,S,D)
    const float* mass_w = (const float*)d_in[1];   // (H,hd)
    const float* v_w    = (const float*)d_in[2];   // (D,D)
    const float* out_w  = (const float*)d_in[3];   // (D,D)
    const float* pos    = (const float*)d_in[4];   // (S,8)
    float* out = (float*)d_out;

    char* ws = (char*)d_ws;
    size_t off = 0;
    auto alloc = [&](size_t bytes) { void* p = ws + off; off = (off + bytes + 255) & ~(size_t)255; return p; };
    float*  rinv    = (float*)alloc((size_t)S_ * S_ * 4);       // 16.8 MB
    __bf16* xbf     = (__bf16*)alloc((size_t)ROWS_ * D_ * 2);   //  8.4 MB
    __bf16* vwbf    = (__bf16*)alloc((size_t)D_ * D_ * 2);      //  2.1 MB
    __bf16* owbf    = (__bf16*)alloc((size_t)D_ * D_ * 2);      //  2.1 MB
    __bf16* Vbf     = (__bf16*)alloc((size_t)ROWS_ * D_ * 2);   //  8.4 MB
    __bf16* Hbf     = (__bf16*)alloc((size_t)ROWS_ * D_ * 2);   //  8.4 MB
    float*  masses  = (float*)alloc((size_t)NROW_ * 4);
    float*  rowmax  = (float*)alloc((size_t)NROW_ * 4);
    float*  rowisum = (float*)alloc((size_t)NROW_ * 4);
    (void)ws_size; (void)in_sizes; (void)n_in; (void)out_size;

    // 1) bf16 conversions
    k_cvt<<<(ROWS_ * D_ / 4 + 255) / 256, 256, 0, stream>>>((const float4*)x, xbf, ROWS_ * D_ / 4);
    k_cvt<<<(D_ * D_ / 4 + 255) / 256, 256, 0, stream>>>((const float4*)v_w, vwbf, D_ * D_ / 4);
    k_cvt<<<(D_ * D_ / 4 + 255) / 256, 256, 0, stream>>>((const float4*)out_w, owbf, D_ * D_ / 4);
    // 2) distance table
    k_rinv<<<(S_ * S_) / 256, 256, 0, stream>>>(pos, rinv);
    // 3) masses
    k_masses<<<NROW_ / 8, 256, 0, stream>>>(x, mass_w, masses);
    // 4) V projection (WMMA)
    k_gemm<0><<<(ROWS_ / 128) * (D_ / 128), 256, 0, stream>>>(xbf, vwbf, (void*)Vbf);
    // 5) softmax row stats
    k_stats<<<NROW_ / 8, 256, 0, stream>>>(rinv, masses, rowmax, rowisum);
    // 6) fused attention P@V (WMMA)
    k_attn<<<BH_ * (S_ / 128), 256, 0, stream>>>(rinv, masses, rowmax, rowisum, Vbf, Hbf);
    // 7) output projection (WMMA, f32 out)
    k_gemm<1><<<(ROWS_ / 128) * (D_ / 128), 256, 0, stream>>>(Hbf, owbf, (void*)out);
}
